// FraudSNN_19851338842381
// MI455X (gfx1250) — matrix operand order repaired and checked by reading, compiled, not run on hardware
//
#include <hip/hip_runtime.h>
#include <hip/hip_bf16.h>

typedef __attribute__((ext_vector_type(16))) __bf16 v16bf;
typedef __attribute__((ext_vector_type(8)))  float  v8f;

static constexpr int   T       = 10;
static constexpr int   F       = 256;
static constexpr int   TILE_B  = 16;
static constexpr int   TCH     = 5;            // timesteps staged per LDS chunk
static constexpr int   RS      = TCH * F + 8;  // 1288 bf16 row stride (2576B; %256==16 -> conflict-free)
static constexpr int   A_ELEMS = TILE_B * RS;
static constexpr size_t SMEM_BYTES =
    (size_t)A_ELEMS * sizeof(__bf16) + (size_t)T * TILE_B * sizeof(float);

union AU  { uint4 u[2]; v16bf v; };
union PK4 { __bf16 h[4]; uint2 u; };

__device__ __forceinline__ float sigf(float z) {
  // sigmoid(z) = 1 / (1 + e^{-z})
  return __builtin_amdgcn_rcpf(1.0f + __expf(-z));
}

__global__ __launch_bounds__(512, 1)
void snn_fused(const float* __restrict__ x,  const float* __restrict__ W1,
               const float* __restrict__ b1, const float* __restrict__ W2,
               const float* __restrict__ b2, float* __restrict__ out)
{
  extern __shared__ char smem[];
  __bf16* ldsA  = (__bf16*)smem;
  float*  cur2f = (float*)(smem + (size_t)A_ELEMS * sizeof(__bf16));

  const int  tid   = threadIdx.x;
  const int  lane  = tid & 31;
  const int  wave  = tid >> 5;
  const int  lhalf = lane >> 4;   // 0: lanes 0-15, 1: lanes 16-31
  const int  lcol  = lane & 15;
  const long bRow  = (long)blockIdx.x * TILE_B;

  // ---- per-wave W1 fragments in bf16 (held in VGPRs for all timesteps) ----
  // B operand, k order chosen here (16-bit, 32x16): lane = column n (mod 16); lanes 0-15 hold
  // K=k0..k0+15, lanes 16-31 hold K=k0+16..k0+31, contiguous within the lane.
  const int n0 = wave * 32 + lcol;
  const int n1 = n0 + 16;
  v16bf Bf[2][8];
#pragma unroll
  for (int nt = 0; nt < 2; ++nt) {
    const int n = nt ? n1 : n0;
    const float4* wp = (const float4*)(W1 + (long)n * F);
#pragma unroll
    for (int k = 0; k < 8; ++k) {
      const int f4 = (k * 32 + lhalf * 16) >> 2;  // 16 contiguous floats
      float4 q0 = wp[f4 + 0], q1 = wp[f4 + 1], q2 = wp[f4 + 2], q3 = wp[f4 + 3];
      v16bf b;
      b[ 0]=(__bf16)q0.x; b[ 1]=(__bf16)q0.y; b[ 2]=(__bf16)q0.z; b[ 3]=(__bf16)q0.w;
      b[ 4]=(__bf16)q1.x; b[ 5]=(__bf16)q1.y; b[ 6]=(__bf16)q1.z; b[ 7]=(__bf16)q1.w;
      b[ 8]=(__bf16)q2.x; b[ 9]=(__bf16)q2.y; b[10]=(__bf16)q2.z; b[11]=(__bf16)q2.w;
      b[12]=(__bf16)q3.x; b[13]=(__bf16)q3.y; b[14]=(__bf16)q3.z; b[15]=(__bf16)q3.w;
      Bf[nt][k] = b;
    }
  }
  const float b10 = b1[n0], b11 = b1[n1];
  const float w20 = W2[n0], w21 = W2[n1];

  if (tid < T * TILE_B) cur2f[tid] = 0.0f;

  float m0[8], m1[8];
#pragma unroll
  for (int r = 0; r < 8; ++r) { m0[r] = 0.0f; m1[r] = 0.0f; }

  const float4* xs4 = (const float4*)x;

  for (int tc = 0; tc < T / TCH; ++tc) {
    __syncthreads();  // previous chunk fully consumed (also covers cur2f init)

    // ---- stage 16 rows x 5 timesteps of x into LDS as bf16 (b128 loads) ----
#pragma unroll
    for (int i = 0; i < (TILE_B * TCH * F / 4) / 512; ++i) {  // 10 iters
      const int j   = tid + i * 512;        // float4 index within chunk
      const int row = j / (TCH * F / 4);    // /320
      const int c4  = j - row * (TCH * F / 4);
      float4 vv = xs4[(bRow + row) * (T * F / 4) + tc * (TCH * F / 4) + c4];
      PK4 p;
      p.h[0] = (__bf16)vv.x; p.h[1] = (__bf16)vv.y;
      p.h[2] = (__bf16)vv.z; p.h[3] = (__bf16)vv.w;
      *(uint2*)(ldsA + row * RS + c4 * 4) = p.u;
    }

    if (tc == 0) {
      // prefetch the second 5-timestep chunk while chunk 0 is computed:
      // 16 rows x 5120B = 640 cachelines of 128B
#pragma unroll
      for (int p = 0; p < 2; ++p) {
        const int line = tid + p * 512;
        if (line < TILE_B * 40) {               // 40 lines per row-chunk
          const int row = line / 40;
          const int off = line - row * 40;
          __builtin_prefetch(x + (bRow + row) * (long)(T * F) + TCH * F + off * 32, 0, 1);
        }
      }
    }
    __syncthreads();

    for (int tt = 0; tt < TCH; ++tt) {
      const int t = tc * TCH + tt;

      v8f c0, c1;
#pragma unroll
      for (int r = 0; r < 8; ++r) { c0[r] = b10; c1[r] = b11; }

      // A operand, same k order as B (16-bit, 16x32): lane m = lane&15; lanes 0-15 hold
      // K=f0..f0+15 (f0 = 32k), lanes 16-31 K=f0+16..f0+31, contiguous.
      const __bf16* arow = ldsA + lcol * RS + tt * F + (lhalf << 4);

      // software-pipelined A-fragment loads: issue k+1's ds_loads before
      // consuming k's, so the compiler can use s_wait_dscnt 2 instead of 0.
      AU a_cur, a_nxt;
      a_cur.u[0] = *(const uint4*)(arow + 0);
      a_cur.u[1] = *(const uint4*)(arow + 8);
#pragma unroll
      for (int k = 0; k < 8; ++k) {
        if (k < 7) {
          a_nxt.u[0] = *(const uint4*)(arow + (k + 1) * 32);
          a_nxt.u[1] = *(const uint4*)(arow + (k + 1) * 32 + 8);
        }
        c0 = __builtin_amdgcn_wmma_f32_16x16x32_bf16(
                 false, a_cur.v, false, Bf[0][k], (short)0, c0, false, false);
        c1 = __builtin_amdgcn_wmma_f32_16x16x32_bf16(
                 false, a_cur.v, false, Bf[1][k], (short)0, c1, false, false);
        a_cur = a_nxt;
      }

      // ---- LIF layer 1 on the C tile + layer-2 partial dot ----
      // C layout: reg r, lanes 0-15 -> row r; lanes 16-31 -> row r+8
#pragma unroll
      for (int r = 0; r < 8; ++r) {
        m0[r] = 0.9f * m0[r] + c0[r];
        float s0 = sigf(10.0f * (m0[r] - 1.0f));
        m0[r] -= s0;
        m1[r] = 0.9f * m1[r] + c1[r];
        float s1 = sigf(10.0f * (m1[r] - 1.0f));
        m1[r] -= s1;
        float rp = s0 * w20 + s1 * w21;
        rp += __shfl_xor(rp, 1, 32);
        rp += __shfl_xor(rp, 2, 32);
        rp += __shfl_xor(rp, 4, 32);
        rp += __shfl_xor(rp, 8, 32);   // halves reduce independently
        if (lcol == 0)
          atomicAdd(&cur2f[t * TILE_B + (lhalf << 3) + r], rp);
      }
    }
  }

  __syncthreads();

  // ---- layer-2 LIF recurrence + readout (16 rows, one lane each) ----
  if (tid < TILE_B) {
    const float bb2 = b2[0];
    float mem2 = 0.0f, acc = 0.0f;
#pragma unroll
    for (int t = 0; t < T; ++t) {
      float cur2 = cur2f[t * TILE_B + tid] + bb2;
      mem2 = 0.9f * mem2 + cur2;
      float s = sigf(10.0f * (mem2 - 1.0f));
      mem2 -= s;
      acc += s;
    }
    out[bRow + tid] = sigf(acc * 0.1f);
  }
}

extern "C" void kernel_launch(void* const* d_in, const int* in_sizes, int n_in,
                              void* d_out, int out_size, void* d_ws, size_t ws_size,
                              hipStream_t stream) {
  const float* x  = (const float*)d_in[0];
  const float* W1 = (const float*)d_in[1];
  const float* b1 = (const float*)d_in[2];
  const float* W2 = (const float*)d_in[3];
  const float* b2 = (const float*)d_in[4];
  float* out = (float*)d_out;

  const int B = in_sizes[0] / (T * F);   // 32768
  dim3 grid(B / TILE_B), block(512);
  snn_fused<<<grid, block, SMEM_BYTES, stream>>>(x, W1, b1, W2, b2, out);
}
